// GIN_27212912788333
// MI455X (gfx1250) — compile-verified, matrix operand order checked by reading, not run on hardware
//
#include <hip/hip_runtime.h>
#include <hip/hip_bf16.h>

// ---------------------------------------------------------------------------
// GIN forward on MI455X (gfx1250, wave32).
// GEMMs via v_wmma_f32_16x16x32_bf16 (f32 accumulate), aggregations via
// native global_atomic_add_f32 (inline asm -> no CAS loop; L2-resident).
// ---------------------------------------------------------------------------

typedef __attribute__((ext_vector_type(16))) __bf16 v16bf;
typedef __attribute__((ext_vector_type(8)))  float  v8f;

#define N_NODES 50000
#define N_EDGES 800000

// --------------------------- atomics ---------------------------------------
__device__ __forceinline__ void atomic_add_f32(float* p, float v) {
    // No-return f32 add: tracked with STOREcnt, completes by s_endpgm's
    // implicit wait-idle. Avoids CAS loops for the 307M edge-atomics.
    asm volatile("global_atomic_add_f32 %0, %1, off" :: "v"(p), "v"(v) : "memory");
}

// --------------------------- elementwise -----------------------------------
__global__ __launch_bounds__(256) void scale_copy(const float* __restrict__ x,
                                                  float* __restrict__ z,
                                                  const float* __restrict__ epsP,
                                                  int n) {
    int i = blockIdx.x * blockDim.x + threadIdx.x;
    if (i < n) z[i] = (1.0f + epsP[0]) * x[i];
}

__global__ __launch_bounds__(256) void f32_to_bf16(const float* __restrict__ in,
                                                   __bf16* __restrict__ out,
                                                   int n) {
    int i = blockIdx.x * blockDim.x + threadIdx.x;
    if (i < n) out[i] = (__bf16)in[i];
}

// --------------------------- edge scatter-add ------------------------------
// One wave (32 lanes) per edge; each lane owns CH/32 contiguous channels.
template<int CH>
__global__ __launch_bounds__(256) void scatter_add(const float* __restrict__ X,
                                                   float* __restrict__ Z,
                                                   const int* __restrict__ ei,
                                                   int nEdges) {
    int g    = blockIdx.x * blockDim.x + threadIdx.x;
    int edge = g >> 5;
    int lane = g & 31;
    if (edge >= nEdges) return;
    int src = ei[edge];            // row 0: source
    int dst = ei[nEdges + edge];   // row 1: target
    const float* xs = X + (size_t)src * CH + lane * (CH / 32);
    float*       zd = Z + (size_t)dst * CH + lane * (CH / 32);
#pragma unroll
    for (int f = 0; f < CH / 32; f += 4) {
        float4 v = *reinterpret_cast<const float4*>(xs + f);
        atomic_add_f32(zd + f + 0, v.x);
        atomic_add_f32(zd + f + 1, v.y);
        atomic_add_f32(zd + f + 2, v.z);
        atomic_add_f32(zd + f + 3, v.w);
    }
}

// --------------------------- weight pre-swizzle ----------------------------
// Pack W[K,N] (f32 row-major) into bf16 "fragment-major" tiles so the GEMM
// B-operand is two contiguous 16B loads per lane. K-pattern mirrors the
// documented 16-bit A layout: elems 0..7 -> K=half*8+e, 8..15 -> K=16+half*8+e.
__global__ __launch_bounds__(256) void prep_w(const float* __restrict__ W,
                                              __bf16* __restrict__ out,
                                              int K, int N) {
    int tid = blockIdx.x * blockDim.x + threadIdx.x;
    if (tid >= K * N) return;
    int frag  = tid >> 9;        // 512 elems per fragment (32 lanes x 16)
    int rem   = tid & 511;
    int lane  = rem >> 4;
    int e     = rem & 15;
    int nKt   = K >> 5;
    int ntile = frag / nKt;
    int kt    = frag - ntile * nKt;
    int half  = lane >> 4;
    int n     = ntile * 16 + (lane & 15);
    int k     = kt * 32 + ((e < 8) ? (half * 8 + e) : (16 + half * 8 + (e - 8)));
    out[tid]  = (__bf16)W[k * N + n];
}

// --------------------------- WMMA GEMM -------------------------------------
// C[M,N] = act(A[M,K] @ B[K,N] + bias). M-tile = 16 rows per block (exact:
// 50000 = 3125*16). 4 waves/block, 2 N-tiles per wave (A-fragment reuse).
template<int KDIM, int NDIM, int RELU, int WRITE_BF, int WRITE_F, int WRITE_F2>
__global__ __launch_bounds__(128) void gemm_wmma(const __bf16* __restrict__ A,
                                                 const __bf16* __restrict__ Bp,
                                                 const float* __restrict__ bias,
                                                 __bf16* __restrict__ outB,
                                                 float* __restrict__ outF,
                                                 float* __restrict__ outF2) {
    constexpr int nKt = KDIM / 32;
    const int lane = threadIdx.x & 31;
    const int wave = threadIdx.x >> 5;
    const int m0   = blockIdx.x * 16;
    const int nt0  = blockIdx.y * 8 + wave * 2;   // this wave's first N-tile
    const int half = lane >> 4;
    const int mrow = m0 + (lane & 15);

    const __bf16* Arow = A + (size_t)mrow * KDIM;

    v8f acc0 = {};
    v8f acc1 = {};
#pragma unroll
    for (int kt = 0; kt < nKt; ++kt) {
        // A fragment: row mrow, K = kb..kb+7 and kb+16..kb+23  (two 16B loads)
        v16bf a;
        uint4* ap = reinterpret_cast<uint4*>(&a);
        const int kb = kt * 32 + half * 8;
        ap[0] = *reinterpret_cast<const uint4*>(Arow + kb);
        ap[1] = *reinterpret_cast<const uint4*>(Arow + kb + 16);

        // B fragments: pre-swizzled, 32 contiguous bytes per lane
        const __bf16* bp0 = Bp + (((size_t)nt0       * nKt + kt) * 32 + lane) * 16;
        const __bf16* bp1 = Bp + (((size_t)(nt0 + 1) * nKt + kt) * 32 + lane) * 16;
        v16bf b0, b1;
        uint4* b0p = reinterpret_cast<uint4*>(&b0);
        uint4* b1p = reinterpret_cast<uint4*>(&b1);
        b0p[0] = reinterpret_cast<const uint4*>(bp0)[0];
        b0p[1] = reinterpret_cast<const uint4*>(bp0)[1];
        b1p[0] = reinterpret_cast<const uint4*>(bp1)[0];
        b1p[1] = reinterpret_cast<const uint4*>(bp1)[1];

        acc0 = __builtin_amdgcn_wmma_f32_16x16x32_bf16(false, a, false, b0,
                                                       (short)0, acc0, false, false);
        acc1 = __builtin_amdgcn_wmma_f32_16x16x32_bf16(false, a, false, b1,
                                                       (short)0, acc1, false, false);
    }

    // Epilogue: C/D layout -> VGPR r holds (M = m0 + r + 8*half, N = lane&15)
    const int ncol0 = nt0 * 16 + (lane & 15);
    const int ncol1 = ncol0 + 16;
    const float bia0 = bias[ncol0];
    const float bia1 = bias[ncol1];
#pragma unroll
    for (int r = 0; r < 8; ++r) {
        const int m = m0 + r + 8 * half;
        float v0 = acc0[r] + bia0;
        float v1 = acc1[r] + bia1;
        if (RELU) { v0 = fmaxf(v0, 0.0f); v1 = fmaxf(v1, 0.0f); }
        const size_t i0 = (size_t)m * NDIM + ncol0;
        const size_t i1 = (size_t)m * NDIM + ncol1;
        if (WRITE_BF) { outB[i0] = (__bf16)v0; outB[i1] = (__bf16)v1; }
        if (WRITE_F)  { outF[i0] = v0;         outF[i1] = v1; }
        if (WRITE_F2) { outF2[i0] = v0;        outF2[i1] = v1; }
    }
}

// --------------------------- launcher --------------------------------------
extern "C" void kernel_launch(void* const* d_in, const int* in_sizes, int n_in,
                              void* d_out, int out_size, void* d_ws, size_t ws_size,
                              hipStream_t stream) {
    const float* x    = (const float*)d_in[0];
    const int*   ei   = (const int*)  d_in[1];
    const float* W1a  = (const float*)d_in[2];
    const float* b1a  = (const float*)d_in[3];
    const float* W1b  = (const float*)d_in[4];
    const float* b1b  = (const float*)d_in[5];
    const float* eps1 = (const float*)d_in[6];
    const float* W2   = (const float*)d_in[7];
    const float* b2   = (const float*)d_in[8];
    const float* W3   = (const float*)d_in[9];
    const float* b3   = (const float*)d_in[10];
    (void)in_sizes; (void)n_in; (void)out_size; (void)ws_size;

    // Workspace layout (~155 MB; everything L2-resident on MI455X's 192MB L2)
    char* ws = (char*)d_ws;
    __bf16* w1a = (__bf16*)(ws + 0);        // 128*256*2 = 64 KB
    __bf16* w1b = (__bf16*)(ws + 65536);    // 256*256*2 = 128 KB
    __bf16* w2  = (__bf16*)(ws + 196608);   // 128 KB
    __bf16* w3  = (__bf16*)(ws + 327680);   // 64 KB
    size_t off = 393216;
    __bf16* B1 = (__bf16*)(ws + off); off += (size_t)N_NODES * 256 * 2; // act bf16
    __bf16* B2 = (__bf16*)(ws + off); off += (size_t)N_NODES * 256 * 2; // act bf16
    float*  B3 = (float*) (ws + off); off += (size_t)N_NODES * 256 * 4; // f32 buf
    float*  B4 = (float*) (ws + off);                                   // f32 buf

    // Pre-swizzle weights to bf16 fragment layout (deterministic each call)
    prep_w<<<(128 * 256) / 256, 256, 0, stream>>>(W1a, w1a, 128, 256);
    prep_w<<<(256 * 256) / 256, 256, 0, stream>>>(W1b, w1b, 256, 256);
    prep_w<<<(256 * 256) / 256, 256, 0, stream>>>(W2,  w2,  256, 256);
    prep_w<<<(256 * 128) / 256, 256, 0, stream>>>(W3,  w3,  256, 128);

    // conv1 aggregation: z1 = (1+eps)*x ; z1[dst] += x[src]
    scale_copy<<<(N_NODES * 128) / 256, 256, 0, stream>>>(x, B3, eps1, N_NODES * 128);
    scatter_add<128><<<(N_EDGES * 32) / 256, 256, 0, stream>>>(x, B3, ei, N_EDGES);
    f32_to_bf16<<<(N_NODES * 128) / 256, 256, 0, stream>>>(B3, B1, N_NODES * 128);

    // mlp1: h = relu(relu(z1 @ W1a + b1a) @ W1b + b1b)
    gemm_wmma<128, 256, 1, 1, 0, 0><<<dim3(3125, 2), 128, 0, stream>>>(
        B1, w1a, b1a, B2, nullptr, nullptr);
    // dual f32 store: B3 = pristine h (gather source), B4 = z2 accumulator init
    gemm_wmma<256, 256, 1, 0, 1, 1><<<dim3(3125, 2), 128, 0, stream>>>(
        B2, w1b, b1b, nullptr, B3, B4);

    // conv2 aggregation: z2 = h + sum h[src]
    scatter_add<256><<<(N_EDGES * 32) / 256, 256, 0, stream>>>(B3, B4, ei, N_EDGES);
    f32_to_bf16<<<(N_NODES * 256) / 256, 256, 0, stream>>>(B4, B1, N_NODES * 256);

    // mlp2 + final linear
    gemm_wmma<256, 256, 1, 1, 0, 0><<<dim3(3125, 2), 128, 0, stream>>>(
        B1, w2, b2, B2, nullptr, nullptr);
    gemm_wmma<256, 128, 0, 0, 1, 0><<<dim3(3125, 1), 128, 0, stream>>>(
        B2, w3, b3, nullptr, (float*)d_out, nullptr);
}
